// MultiHeadedAttention_79250736546673
// MI455X (gfx1250) — compile-verified, matrix operand order checked by reading, not run on hardware
//
#include <hip/hip_runtime.h>
#include <hip/hip_bf16.h>

// ---------------------------------------------------------------------------
// MultiHeadedAttention for MI455X (gfx1250, wave32, WMMA).
//   B=2, S=4096, D_MODEL=4096, N_HEADS=2, HEAD_DIM=2048.
// 4 big GEMMs (split-bf16 on v_wmma_f32_16x16x32_bf16) + tiny per-position
// RoPE / 2x2-head-attention kernel. ws usage: 4 * 8192*4096 fp32 = 512 MB.
// Round 2: software-pipelined tile loop (register prefetch of next global
// tile overlapped with WMMA compute) to hide HBM/L2 latency in-block.
// ---------------------------------------------------------------------------

typedef __bf16 bf16x4 __attribute__((ext_vector_type(4)));
typedef __bf16 bf16x8 __attribute__((ext_vector_type(8)));
typedef __bf16 v16bf  __attribute__((ext_vector_type(16)));
typedef float  v8f    __attribute__((ext_vector_type(8)));

#define BM 128
#define BN 128
#define BK 32
#define LDS_STRIDE 40  // elems; 80B rows -> 16B-aligned frag reads, no bank conflicts

// Load one 16x32 bf16 WMMA operand fragment from an LDS tile stored row-major
// with LDS_STRIDE element pitch. Per CDNA5 ISA layout (A and B operands are
// loaded identically here since B := row n of W == column n of W^T):
//   lane&15  -> matrix row (A: M, B: N)
//   lane>>4  -> K base 0 or 8 ; VGPRs 0-3 hold K base..base+7,
//   VGPRs 4-7 hold K base+16..base+23  => two contiguous 8-elem chunks.
__device__ inline v16bf load_frag(const __bf16* base, int row0, int lane) {
  const int r  = row0 + (lane & 15);
  const int kb = (lane >> 4) * 8;
  union { v16bf v; bf16x8 h[2]; } u;
  u.h[0] = *(const bf16x8*)(base + r * LDS_STRIDE + kb);
  u.h[1] = *(const bf16x8*)(base + r * LDS_STRIDE + kb + 16);
  return u.v;
}

// C[M x N] = A[M x K] * W[N x K]^T   (einsum 'md,nd->mn'), fp32 in/out.
// Split-bf16: each fp32 value -> hi + lo bf16; product uses 3 WMMAs:
// ahi*bhi + ahi*blo + alo*bhi, accumulated in fp32.
__global__ __launch_bounds__(256)
void gemm_xWT_bf16split_kernel(const float* __restrict__ A,
                               const float* __restrict__ W,
                               float* __restrict__ C,
                               int M, int N, int K) {
  __shared__ __bf16 sAhi[BM * LDS_STRIDE];
  __shared__ __bf16 sAlo[BM * LDS_STRIDE];
  __shared__ __bf16 sBhi[BN * LDS_STRIDE];
  __shared__ __bf16 sBlo[BN * LDS_STRIDE];

  const int tid  = threadIdx.x;
  const int lane = tid & 31;
  const int wave = tid >> 5;
  const int wm   = (wave & 3) * 32;  // wave's row offset in block tile
  const int wn   = (wave >> 2) * 64; // wave's col offset in block tile

  const int m0 = blockIdx.y * BM;
  const int n0 = blockIdx.x * BN;

  // Per-thread tile-load coordinates (4 float4 per matrix per step).
  const int lr = tid >> 3;        // base row 0..31 (stepped by 32 per chunk)
  const int lc = (tid & 7) * 4;   // col 0,4,...,28

  v8f acc[2][4] = {};

  const int nsteps = K / BK;
  float4 pa[4], pb[4];

  // Prologue: prefetch k-step 0 into registers.
#pragma unroll
  for (int i = 0; i < 4; ++i) {
    const int r = lr + 32 * i;
    pa[i] = *(const float4*)(A + (size_t)(m0 + r) * K + lc);
    pb[i] = *(const float4*)(W + (size_t)(n0 + r) * K + lc);
  }

  for (int step = 0; step < nsteps; ++step) {
    __syncthreads();  // previous step's LDS reads complete before overwrite
    // Convert prefetched fp32 to hi/lo bf16 and stage into LDS.
#pragma unroll
    for (int i = 0; i < 4; ++i) {
      const int r = lr + 32 * i;
      const float fav[4] = {pa[i].x, pa[i].y, pa[i].z, pa[i].w};
      const float fbv[4] = {pb[i].x, pb[i].y, pb[i].z, pb[i].w};
      bf16x4 ah, al, bh, bl;
#pragma unroll
      for (int u = 0; u < 4; ++u) {
        const __bf16 h = (__bf16)fav[u];
        ah[u] = h;
        al[u] = (__bf16)(fav[u] - (float)h);
        const __bf16 g = (__bf16)fbv[u];
        bh[u] = g;
        bl[u] = (__bf16)(fbv[u] - (float)g);
      }
      *(bf16x4*)(&sAhi[r * LDS_STRIDE + lc]) = ah;
      *(bf16x4*)(&sAlo[r * LDS_STRIDE + lc]) = al;
      *(bf16x4*)(&sBhi[r * LDS_STRIDE + lc]) = bh;
      *(bf16x4*)(&sBlo[r * LDS_STRIDE + lc]) = bl;
    }
    __syncthreads();

    // Issue next step's global loads now; their latency is hidden behind the
    // WMMA compute below (waited on only at the next store phase).
    if (step + 1 < nsteps) {
      const int kn = (step + 1) * BK;
#pragma unroll
      for (int i = 0; i < 4; ++i) {
        const int r = lr + 32 * i;
        pa[i] = *(const float4*)(A + (size_t)(m0 + r) * K + kn + lc);
        pb[i] = *(const float4*)(W + (size_t)(n0 + r) * K + kn + lc);
      }
    }

    v16bf ahi[2], alo[2];
#pragma unroll
    for (int mi = 0; mi < 2; ++mi) {
      ahi[mi] = load_frag(sAhi, wm + mi * 16, lane);
      alo[mi] = load_frag(sAlo, wm + mi * 16, lane);
    }
#pragma unroll
    for (int ni = 0; ni < 4; ++ni) {
      const v16bf bhi = load_frag(sBhi, wn + ni * 16, lane);
      const v16bf blo = load_frag(sBlo, wn + ni * 16, lane);
#pragma unroll
      for (int mi = 0; mi < 2; ++mi) {
        acc[mi][ni] = __builtin_amdgcn_wmma_f32_16x16x32_bf16(
            false, alo[mi], false, bhi, (short)0, acc[mi][ni], false, false);
        acc[mi][ni] = __builtin_amdgcn_wmma_f32_16x16x32_bf16(
            false, ahi[mi], false, blo, (short)0, acc[mi][ni], false, false);
        acc[mi][ni] = __builtin_amdgcn_wmma_f32_16x16x32_bf16(
            false, ahi[mi], false, bhi, (short)0, acc[mi][ni], false, false);
      }
    }
  }

  // Epilogue: C/D layout = VGPR r: lanes 0-15 -> M=r, lanes 16-31 -> M=r+8;
  // N = lane & 15.
#pragma unroll
  for (int mi = 0; mi < 2; ++mi) {
    const int mrow = m0 + wm + mi * 16 + (lane >> 4) * 8;
#pragma unroll
    for (int ni = 0; ni < 4; ++ni) {
      const int ncol = n0 + wn + ni * 16 + (lane & 15);
#pragma unroll
      for (int r = 0; r < 8; ++r) {
        C[(size_t)(mrow + r) * N + ncol] = acc[mi][ni][r];
      }
    }
  }
}

// Per-position RoPE + 2-head causal attention. One block per (b,s).
// scores are 2x2 across heads; causal => head0 ctx = v0, head1 ctx =
// softmax([q1.k0, q1.k1]/sqrt(2048)) blend of v0,v1 (q/k RoPE'd, v raw).
// ctx is written in the reference's transpose(0,2,1,3).reshape(B,S,D) layout:
//   head0 -> row (b*4096 + s/2),        col (s&1)*2048 + e
//   head1 -> row (b*4096 + 2048 + s/2), col (s&1)*2048 + e
__global__ __launch_bounds__(256)
void rope_headattn_kernel(const float* __restrict__ q,
                          const float* __restrict__ k,
                          const float* __restrict__ v,
                          float* __restrict__ ctx) {
  const int p   = blockIdx.x;     // 0..8191 == b*4096 + s
  const int s   = p & 4095;
  const int tid = threadIdx.x;

  const float* qp = q + (size_t)p * 4096;
  const float* kp = k + (size_t)p * 4096;
  const float* vp = v + (size_t)p * 4096;

  __shared__ float red0[256];
  __shared__ float red1[256];
  __shared__ float a01[2];

  const float LOG_BASE = 9.210340371976184f;  // ln(10000)
  float d0 = 0.0f, d1 = 0.0f;
  for (int j = tid; j < 1024; j += 256) {
    const float inv = expf(-LOG_BASE * (float)j * (1.0f / 1024.0f));
    const float fr  = (float)s * inv;
    const float cs  = cosf(fr);
    const float sn  = sinf(fr);
    // RoPE'd q head1 pair j
    const float q0 = qp[2048 + 2 * j], q1 = qp[2048 + 2 * j + 1];
    const float rq0 = q0 * cs - q1 * sn, rq1 = q1 * cs + q0 * sn;
    // RoPE'd k head0 / head1 pairs
    const float k00 = kp[2 * j], k01 = kp[2 * j + 1];
    const float rk00 = k00 * cs - k01 * sn, rk01 = k01 * cs + k00 * sn;
    const float k10 = kp[2048 + 2 * j], k11 = kp[2048 + 2 * j + 1];
    const float rk10 = k10 * cs - k11 * sn, rk11 = k11 * cs + k10 * sn;
    d0 += rq0 * rk00 + rq1 * rk01;
    d1 += rq0 * rk10 + rq1 * rk11;
  }
  red0[tid] = d0;
  red1[tid] = d1;
  __syncthreads();
  for (int off = 128; off > 0; off >>= 1) {
    if (tid < off) {
      red0[tid] += red0[tid + off];
      red1[tid] += red1[tid + off];
    }
    __syncthreads();
  }
  if (tid == 0) {
    const float scale = 0.022097086912079608f;  // 1/sqrt(2048)
    const float z0 = red0[0] * scale, z1 = red1[0] * scale;
    const float mx = fmaxf(z0, z1);
    const float e0 = expf(z0 - mx), e1 = expf(z1 - mx);
    const float rinv = 1.0f / (e0 + e1);
    a01[0] = e0 * rinv;
    a01[1] = e1 * rinv;
  }
  __syncthreads();
  const float a0 = a01[0], a1 = a01[1];

  const size_t b  = (size_t)(p >> 12);
  const size_t t  = (size_t)(s >> 1);
  const size_t cb = (size_t)(s & 1) * 2048;
  float* row0 = ctx + (b * 4096 + t) * 4096 + cb;           // head0 dest
  float* row1 = ctx + (b * 4096 + 2048 + t) * 4096 + cb;    // head1 dest
  for (int e = tid; e < 2048; e += 256) {
    const float v0 = vp[e];
    const float v1 = vp[2048 + e];
    row0[e] = v0;
    row1[e] = a0 * v0 + a1 * v1;
  }
}

extern "C" void kernel_launch(void* const* d_in, const int* in_sizes, int n_in,
                              void* d_out, int out_size, void* d_ws, size_t ws_size,
                              hipStream_t stream) {
  (void)in_sizes; (void)n_in; (void)out_size; (void)ws_size;
  const float* x  = (const float*)d_in[0];
  const float* Wq = (const float*)d_in[1];
  const float* Wk = (const float*)d_in[2];
  const float* Wv = (const float*)d_in[3];
  const float* Wo = (const float*)d_in[4];
  float* out = (float*)d_out;

  const int M = 8192;   // B * S
  const int N = 4096;   // D_MODEL
  const int K = 4096;   // D_MODEL
  const size_t PLANE = (size_t)M * (size_t)N;

  float* qws = (float*)d_ws;
  float* kws = qws + PLANE;
  float* vws = kws + PLANE;
  float* cws = vws + PLANE;   // ctx in scrambled (transpose+reshape) layout

  dim3 grid(N / BN, M / BM);
  dim3 blk(256);
  hipLaunchKernelGGL(gemm_xWT_bf16split_kernel, grid, blk, 0, stream, x, Wq, qws, M, N, K);
  hipLaunchKernelGGL(gemm_xWT_bf16split_kernel, grid, blk, 0, stream, x, Wk, kws, M, N, K);
  hipLaunchKernelGGL(gemm_xWT_bf16split_kernel, grid, blk, 0, stream, x, Wv, vws, M, N, K);
  hipLaunchKernelGGL(rope_headattn_kernel, dim3(M), blk, 0, stream, qws, kws, vws, cws);
  hipLaunchKernelGGL(gemm_xWT_bf16split_kernel, grid, blk, 0, stream, cws, Wo, out, M, N, K);
}
